// RelativeMultiheadSelfAttention_49005576847876
// MI455X (gfx1250) — matrix operand order repaired and checked by reading, not compiled, not run on hardware
//
#include <hip/hip_runtime.h>

#define D_MODEL 1024
#define NHEADS  16
#define HD      64
#define BATCH   2
#define SEQ     2048

typedef __attribute__((ext_vector_type(16))) __bf16 v16bf;
typedef __attribute__((ext_vector_type(8)))  __bf16 v8bf;
typedef __attribute__((ext_vector_type(8)))  float  v8f;

#define CAT16(lo, hi) __builtin_shufflevector(lo, hi, 0,1,2,3,4,5,6,7,8,9,10,11,12,13,14,15)

__device__ __forceinline__ __bf16 f2bf(float f) {
    unsigned u = __builtin_bit_cast(unsigned, f);
    u += 0x7FFFu + ((u >> 16) & 1u);            // round-to-nearest-even
    unsigned short s = (unsigned short)(u >> 16);
    return __builtin_bit_cast(__bf16, s);
}

__device__ __forceinline__ v8f wmma_bf16(v16bf a, v16bf b, v8f c) {
    return __builtin_amdgcn_wmma_f32_16x16x32_bf16(
        /*neg_a=*/false, a, /*neg_b=*/false, b,
        /*c_mod=*/(short)0, c, /*reuse_a=*/false, /*reuse_b=*/false);
}

// ---------------------------------------------------------------------------
// One-shot fp32 -> bf16 pack (x and the four weight matrices).
// n must be a multiple of 2048 (it is: 4M and 1M element arrays).
// ---------------------------------------------------------------------------
__global__ __launch_bounds__(256) void pack_bf16_kernel(
    const float* __restrict__ in, __bf16* __restrict__ out)
{
    size_t i = ((size_t)blockIdx.x * 256 + threadIdx.x) * 8;
    float4 a = *(const float4*)(in + i);
    float4 b = *(const float4*)(in + i + 4);
    v8bf o;
    o[0] = f2bf(a.x); o[1] = f2bf(a.y); o[2] = f2bf(a.z); o[3] = f2bf(a.w);
    o[4] = f2bf(b.x); o[5] = f2bf(b.y); o[6] = f2bf(b.z); o[7] = f2bf(b.w);
    *(v8bf*)(out + i) = o;
}

// ---------------------------------------------------------------------------
// Projection GEMM (all-bf16 operands): y[m,n] = sum_k A[m,k]*W[n,k] + bias[n]
// Block = 256 threads = 8 waves; tile = 64 rows x 128 cols.
// Each wave: one 16-col strip, 4 row-tiles -> 4 WMMA per K-step reusing B.
// mode 0/1: bf16 out [B,H,S,HD] (Q/K); mode 2: bf16 [B,H,HD,S] (V^T);
// mode 3: fp32 out [4096,1024] (O-projection).
// ---------------------------------------------------------------------------
__global__ __launch_bounds__(256) void gemm_bf16_kernel(
    const __bf16* __restrict__ A,    // [4096, 1024] bf16 row-major
    const __bf16* __restrict__ Wb,   // [1024, 1024] bf16 (torch [out,in])
    const float*  __restrict__ bias,
    __bf16* __restrict__ out_bf,
    float*  __restrict__ out_f,
    int mode)
{
    __shared__ __bf16 lds_a[64 * 32];   // 4 KB A tile

    const int lane = threadIdx.x & 31;
    const int wave = threadIdx.x >> 5;
    const int hl   = lane >> 4;
    const int ln   = lane & 15;

    const int m0 = blockIdx.x * 64;
    const int n  = (blockIdx.y * 8 + wave) * 16 + ln;

    v8f acc[4] = {{}, {}, {}, {}};

    for (int k0 = 0; k0 < D_MODEL; k0 += 32) {
        __syncthreads();
        {   // stage A tile 64x32 bf16: 2048 elems, 8 per thread, one b128 each
            int e = threadIdx.x * 8;
            int r = e >> 5, c = e & 31;
            const __bf16* src = &A[(size_t)(m0 + r) * D_MODEL + k0 + c];
            *(v8bf*)&lds_a[e] = *(const v8bf*)src;
            if (k0 + 32 < D_MODEL)
                __builtin_prefetch(src + 32, 0, 0);   // next K tile of A
        }
        __syncthreads();

        // B fragment: element i -> k = k0 + hl*16 + i (contiguous bf16)
        const __bf16* wp = &Wb[(size_t)n * D_MODEL + k0 + hl * 16];
        v16bf bfrag = CAT16(*(const v8bf*)wp, *(const v8bf*)(wp + 8));
        if (k0 + 32 < D_MODEL)
            __builtin_prefetch(wp + 32, 0, 0);        // next K tile of W

        #pragma unroll
        for (int mt = 0; mt < 4; ++mt) {
            const __bf16* ap = &lds_a[(mt * 16 + ln) * 32 + hl * 16];
            v16bf afrag = CAT16(*(const v8bf*)ap, *(const v8bf*)(ap + 8));
            acc[mt] = wmma_bf16(afrag, bfrag, acc[mt]);
        }
    }

    const float bn = bias[n];
    #pragma unroll
    for (int mt = 0; mt < 4; ++mt) {
        #pragma unroll
        for (int j = 0; j < 8; ++j) {
            int row = m0 + mt * 16 + j + hl * 8;      // global row in [0,4096)
            float v = acc[mt][j] + bn;
            if (mode == 3) {
                out_f[(size_t)row * D_MODEL + n] = v;
            } else {
                int b  = row >> 11;
                int s  = row & (SEQ - 1);
                int h  = n >> 6;
                int hd = n & 63;
                if (mode == 2) {                       // V: [B,H,HD,S]
                    out_bf[(((size_t)(b * NHEADS + h) * HD + hd) * SEQ) + s] = f2bf(v);
                } else {                               // Q/K: [B,H,S,HD]
                    out_bf[(((size_t)(b * NHEADS + h) * SEQ + s) * HD) + hd] = f2bf(v);
                }
            }
        }
    }
}

// ---------------------------------------------------------------------------
// Flash attention with relative bias + mask.
// Grid: (SEQ/128, B*H); block 256 = 8 waves, each owns a 16-row query tile.
// K/V tiles staged once per block into LDS (8x reuse across waves).
// ---------------------------------------------------------------------------
__global__ __launch_bounds__(256) void attn_kernel(
    const __bf16* __restrict__ qw,       // [B,H,S,HD]
    const __bf16* __restrict__ kw,       // [B,H,S,HD]
    const __bf16* __restrict__ vt,       // [B,H,HD,S]
    const float*  __restrict__ rel_bias, // [H,S,S]
    const float*  __restrict__ mask,     // [S,S]
    __bf16* __restrict__ ctx)            // [B*S, D_MODEL] bf16
{
    __shared__ __bf16 kbuf[32 * 64];      // key-tile: [key][hd]
    __shared__ __bf16 vbuf[64 * 32];      // V^T tile: [hd][key]
    __shared__ __bf16 plds[8][16 * 32];   // per-wave P reshape buffer

    const int lane = threadIdx.x & 31;
    const int wave = threadIdx.x >> 5;
    const int hl   = lane >> 4;
    const int ln   = lane & 15;

    const int bh = blockIdx.y;
    const int b  = bh >> 4;
    const int h  = bh & 15;
    const int q0 = (blockIdx.x * 8 + wave) * 16;

    const __bf16* qp = qw + (size_t)bh * SEQ * HD;
    const __bf16* kp = kw + (size_t)bh * SEQ * HD;
    const __bf16* vp = vt + (size_t)bh * HD * SEQ;
    const float*  rb = rel_bias + (size_t)h * SEQ * SEQ;

    // Q A-fragments for contraction chunks d=[0,32) and d=[32,64)
    const __bf16* qrp = qp + (size_t)(q0 + ln) * HD;
    v16bf qa0 = CAT16(*(const v8bf*)(qrp + hl * 16),      *(const v8bf*)(qrp + hl * 16 + 8));
    v16bf qa1 = CAT16(*(const v8bf*)(qrp + 32 + hl * 16), *(const v8bf*)(qrp + 32 + hl * 16 + 8));

    float mi[8], li[8];
    #pragma unroll
    for (int j = 0; j < 8; ++j) { mi[j] = -1e30f; li[j] = 0.f; }
    v8f accv[4] = {{}, {}, {}, {}};
    const float scale = 0.125f;   // 1/sqrt(64)

    __bf16* pw = &plds[wave][0];
    const int e8 = threadIdx.x * 8;        // cooperative-load element base

    for (int kt = 0; kt < SEQ / 32; ++kt) {
        const int kb = kt * 32;

        __syncthreads();                    // prior-iter LDS reads done
        {   // stage K tile [32 keys][64 hd]: 2048 elems, 8/thread
            int r = e8 >> 6, c = e8 & 63;
            const __bf16* src = &kp[(size_t)(kb + r) * HD + c];
            *(v8bf*)&kbuf[e8] = *(const v8bf*)src;
            if (kb + 32 < SEQ)
                __builtin_prefetch(src + 32 * HD, 0, 0);
        }
        {   // stage V^T tile [64 hd][32 keys]: 2048 elems, 8/thread
            int r = e8 >> 5, c = e8 & 31;
            const __bf16* src = &vp[(size_t)r * SEQ + kb + c];
            *(v8bf*)&vbuf[e8] = *(const v8bf*)src;
            if (kb + 32 < SEQ)
                __builtin_prefetch(src + 32, 0, 0);
        }
        __syncthreads();

        // ---- scores: S = Q @ K^T  (two 16-key N-tiles x two 32-deep chunks)
        v8f s0 = {}, s1 = {};
        #pragma unroll
        for (int t = 0; t < 2; ++t) {
            const __bf16* kr = &kbuf[(t * 16 + ln) * HD];
            v16bf bk0 = CAT16(*(const v8bf*)(kr + hl * 16),
                              *(const v8bf*)(kr + hl * 16 + 8));
            v16bf bk1 = CAT16(*(const v8bf*)(kr + 32 + hl * 16),
                              *(const v8bf*)(kr + 32 + hl * 16 + 8));
            if (t == 0) { s0 = wmma_bf16(qa0, bk0, s0); s0 = wmma_bf16(qa1, bk1, s0); }
            else        { s1 = wmma_bf16(qa0, bk0, s1); s1 = wmma_bf16(qa1, bk1, s1); }
        }

        // ---- fused scale + rel_bias + mask, online softmax
        const int col0 = kb + ln;
        float p0[8], p1[8], alpha[8];
        #pragma unroll
        for (int j = 0; j < 8; ++j) {
            int row = q0 + j + hl * 8;
            size_t roff = (size_t)row * SEQ + col0;
            float v0 = s0[j] * scale + rb[roff]      + mask[roff];
            float v1 = s1[j] * scale + rb[roff + 16] + mask[roff + 16];
            float rm = fmaxf(v0, v1);
            rm = fmaxf(rm, __shfl_xor(rm, 1, 32));
            rm = fmaxf(rm, __shfl_xor(rm, 2, 32));
            rm = fmaxf(rm, __shfl_xor(rm, 4, 32));
            rm = fmaxf(rm, __shfl_xor(rm, 8, 32));
            float mn = fmaxf(mi[j], rm);
            p0[j] = __expf(v0 - mn);
            p1[j] = __expf(v1 - mn);
            float rs = p0[j] + p1[j];
            rs += __shfl_xor(rs, 1, 32);
            rs += __shfl_xor(rs, 2, 32);
            rs += __shfl_xor(rs, 4, 32);
            rs += __shfl_xor(rs, 8, 32);
            alpha[j] = __expf(mi[j] - mn);
            li[j] = li[j] * alpha[j] + rs;
            mi[j] = mn;
        }
        #pragma unroll
        for (int j = 0; j < 8; ++j) {
            accv[0][j] *= alpha[j]; accv[1][j] *= alpha[j];
            accv[2][j] *= alpha[j]; accv[3][j] *= alpha[j];
        }

        // ---- reshape P (C-layout) -> A-fragment via per-wave LDS
        #pragma unroll
        for (int j = 0; j < 8; ++j) {
            int row = j + hl * 8;
            pw[row * 32 + ln]      = f2bf(p0[j]);
            pw[row * 32 + 16 + ln] = f2bf(p1[j]);
        }
        asm volatile("s_wait_dscnt 0" ::: "memory");   // cross-lane LDS RAW
        v16bf pa = CAT16(*(const v8bf*)&pw[ln * 32 + hl * 16],
                         *(const v8bf*)&pw[ln * 32 + hl * 16 + 8]);

        // ---- ctx += P @ V   (contraction = key index, contiguous in vbuf)
        #pragma unroll
        for (int t = 0; t < 4; ++t) {
            const __bf16* vr = &vbuf[(t * 16 + ln) * 32 + hl * 16];
            v16bf bv = CAT16(*(const v8bf*)vr, *(const v8bf*)(vr + 8));
            accv[t] = wmma_bf16(pa, bv, accv[t]);
        }
    }

    // ---- epilogue: normalize, write ctx bf16 [B*S, D_MODEL]
    #pragma unroll
    for (int j = 0; j < 8; ++j) {
        int srow = q0 + j + hl * 8;
        float inv = 1.0f / li[j];
        size_t base = ((size_t)b * SEQ + srow) * D_MODEL + h * HD + ln;
        #pragma unroll
        for (int t = 0; t < 4; ++t)
            ctx[base + t * 16] = f2bf(accv[t][j] * inv);
    }
}

// ---------------------------------------------------------------------------
extern "C" void kernel_launch(void* const* d_in, const int* in_sizes, int n_in,
                              void* d_out, int out_size, void* d_ws, size_t ws_size,
                              hipStream_t stream) {
    const float* x    = (const float*)d_in[0];
    const float* rbias= (const float*)d_in[1];
    const float* mask = (const float*)d_in[2];
    const float* Wq   = (const float*)d_in[3];
    const float* bq   = (const float*)d_in[4];
    const float* Wk   = (const float*)d_in[5];
    const float* bk   = (const float*)d_in[6];
    const float* Wv   = (const float*)d_in[7];
    const float* bv   = (const float*)d_in[8];
    const float* Wo   = (const float*)d_in[9];
    const float* bo   = (const float*)d_in[10];

    const size_t NTOK   = (size_t)BATCH * SEQ;              // 4096
    const size_t X_ELEM = NTOK * D_MODEL;                   // 4M
    const size_t W_ELEM = (size_t)D_MODEL * D_MODEL;        // 1M
    const size_t QKV_E  = (size_t)BATCH * NHEADS * SEQ * HD;// 4M

    char* ws = (char*)d_ws;
    __bf16* xbf  = (__bf16*)ws;                       ws += X_ELEM * 2;   // 8 MB
    __bf16* wbf0 = (__bf16*)ws;                       ws += W_ELEM * 2;   // 2 MB
    __bf16* wbf1 = (__bf16*)ws;                       ws += W_ELEM * 2;
    __bf16* wbf2 = (__bf16*)ws;                       ws += W_ELEM * 2;
    __bf16* wbf3 = (__bf16*)ws;                       ws += W_ELEM * 2;
    __bf16* qws  = (__bf16*)ws;                       ws += QKV_E * 2;    // 8 MB
    __bf16* kws  = (__bf16*)ws;                       ws += QKV_E * 2;
    __bf16* vws  = (__bf16*)ws;                       ws += QKV_E * 2;
    __bf16* ctxb = (__bf16*)ws;                       ws += X_ELEM * 2;   // 8 MB

    dim3 blk(256);

    pack_bf16_kernel<<<dim3(X_ELEM / 2048), blk, 0, stream>>>(x,  xbf);
    pack_bf16_kernel<<<dim3(W_ELEM / 2048), blk, 0, stream>>>(Wq, wbf0);
    pack_bf16_kernel<<<dim3(W_ELEM / 2048), blk, 0, stream>>>(Wk, wbf1);
    pack_bf16_kernel<<<dim3(W_ELEM / 2048), blk, 0, stream>>>(Wv, wbf2);
    pack_bf16_kernel<<<dim3(W_ELEM / 2048), blk, 0, stream>>>(Wo, wbf3);

    dim3 gproj(NTOK / 64, D_MODEL / 128);     // 64 x 8 blocks
    gemm_bf16_kernel<<<gproj, blk, 0, stream>>>(xbf, wbf0, bq, qws, nullptr, 0);
    gemm_bf16_kernel<<<gproj, blk, 0, stream>>>(xbf, wbf1, bk, kws, nullptr, 1);
    gemm_bf16_kernel<<<gproj, blk, 0, stream>>>(xbf, wbf2, bv, vws, nullptr, 2);

    dim3 gattn(SEQ / 128, BATCH * NHEADS);
    attn_kernel<<<gattn, blk, 0, stream>>>(qws, kws, vws, rbias, mask, ctxb);

    gemm_bf16_kernel<<<gproj, blk, 0, stream>>>(ctxb, wbf3, bo, nullptr, (float*)d_out, 3);
}
